// TemporalMaskedGraphAttention_14053132993207
// MI455X (gfx1250) — matrix operand order repaired and checked by reading, not compiled, not run on hardware
//
#include <hip/hip_runtime.h>
#include <hip/hip_bf16.h>

typedef unsigned int uint;
typedef __attribute__((ext_vector_type(4)))  uint    v4u;
typedef __attribute__((ext_vector_type(16))) __bf16  v16bf;
typedef __attribute__((ext_vector_type(8)))  float   v8f;

union BFrag { v4u u[2]; v16bf v; };

// ---------------------------------------------------------------------------
// f32 -> bf16 conversion (bandwidth-bound pre-pass)
// ---------------------------------------------------------------------------
__global__ void f32_to_bf16_kernel(const float* __restrict__ in,
                                   __hip_bfloat16* __restrict__ out, int n) {
    int i = blockIdx.x * blockDim.x + threadIdx.x;
    if (i < n) out[i] = __float2bfloat16(in[i]);
}

// ---------------------------------------------------------------------------
// Tiled bf16 WMMA GEMM:  C[M,N] = A[M,K] * W[N,K]^T + bias[N]
// Block tile 128x128, BK=32, 8 waves (wave32), each wave computes 32x64.
// MODE 0: QKV epilogue -> scatter bf16 into Q[B,H,T,HD], K[B,H,T,HD], Vt[B,H,HD,T]
// MODE 1: f32 epilogue -> outF[M,1024]
// ---------------------------------------------------------------------------
template <int MODE>
__global__ void __launch_bounds__(256)
gemm_bf16_kernel(const __hip_bfloat16* __restrict__ A,
                 const __hip_bfloat16* __restrict__ W,
                 const float* __restrict__ bias,
                 float* __restrict__ outF,
                 __hip_bfloat16* __restrict__ Qb,
                 __hip_bfloat16* __restrict__ Kb,
                 __hip_bfloat16* __restrict__ Vtb) {
    constexpr int K   = 1024;
    constexpr int LDA = 40;                 // padded LDS row stride (bf16 elems), 80B
    __shared__ alignas(16) __hip_bfloat16 sA[128 * LDA];
    __shared__ alignas(16) __hip_bfloat16 sW[128 * LDA];

    const int tid  = threadIdx.x;
    const int lane = tid & 31;
    const int wave = tid >> 5;              // 0..7
    const int wm   = wave >> 1;             // 0..3 -> 32-row M slab
    const int wn   = wave & 1;              // 0..1 -> 64-col N slab
    const int lm   = lane & 15;
    const int hs   = lane >> 4;             // half-select
    const int m0   = blockIdx.x * 128;
    const int n0   = blockIdx.y * 128;

    const v8f vzero = {0.f, 0.f, 0.f, 0.f, 0.f, 0.f, 0.f, 0.f};
    v8f acc[2][4];
#pragma unroll
    for (int mt = 0; mt < 2; ++mt)
#pragma unroll
        for (int nt = 0; nt < 4; ++nt) acc[mt][nt] = vzero;

    const int lrow  = tid >> 2;             // 0..63
    const int lcolB = (tid & 3) * 8;        // bf16 element offset (16B chunks)

    for (int k0 = 0; k0 < K; k0 += 32) {
#pragma unroll
        for (int rr = 0; rr < 2; ++rr) {
            const int row = lrow + rr * 64;
            *(v4u*)(sA + row * LDA + lcolB) =
                *(const v4u*)(A + (size_t)(m0 + row) * K + k0 + lcolB);
            *(v4u*)(sW + row * LDA + lcolB) =
                *(const v4u*)(W + (size_t)(n0 + row) * K + k0 + lcolB);
        }
        __syncthreads();

        BFrag af[2], bf[4];
#pragma unroll
        for (int mt = 0; mt < 2; ++mt) {
            const __hip_bfloat16* p = sA + (wm * 32 + mt * 16 + lm) * LDA + hs * 8;
            af[mt].u[0] = *(const v4u*)p;
            af[mt].u[1] = *(const v4u*)(p + 16);
        }
#pragma unroll
        for (int nt = 0; nt < 4; ++nt) {
            const __hip_bfloat16* p = sW + (wn * 64 + nt * 16 + lm) * LDA + hs * 8;
            bf[nt].u[0] = *(const v4u*)p;
            bf[nt].u[1] = *(const v4u*)(p + 16);
        }
#pragma unroll
        for (int mt = 0; mt < 2; ++mt)
#pragma unroll
            for (int nt = 0; nt < 4; ++nt)
                acc[mt][nt] = __builtin_amdgcn_wmma_f32_16x16x32_bf16(
                    false, af[mt].v, false, bf[nt].v, (short)0, acc[mt][nt],
                    false, false);
        __syncthreads();
    }

#pragma unroll
    for (int mt = 0; mt < 2; ++mt)
#pragma unroll
        for (int nt = 0; nt < 4; ++nt)
#pragma unroll
            for (int r = 0; r < 8; ++r) {
                const int mg = m0 + wm * 32 + mt * 16 + r + 8 * hs;
                const int ng = n0 + wn * 64 + nt * 16 + lm;
                const float v = acc[mt][nt][r] + bias[ng];
                if (MODE == 1) {
                    outF[(size_t)mg * 1024 + ng] = v;
                } else {
                    const int comp = ng >> 10;          // 0=Q 1=K 2=V
                    const int c    = ng & 1023;
                    const int h    = c >> 6;
                    const int d    = c & 63;
                    const int b    = mg >> 10;
                    const int t    = mg & 1023;
                    const int bh   = b * 16 + h;
                    const __hip_bfloat16 bv = __float2bfloat16(v);
                    if (comp == 0)      Qb [((size_t)bh * 1024 + t) * 64 + d] = bv;
                    else if (comp == 1) Kb [((size_t)bh * 1024 + t) * 64 + d] = bv;
                    else                Vtb[((size_t)bh * 64 + d) * 1024 + t] = bv;
                }
            }
}

// ---------------------------------------------------------------------------
// Flash-style masked attention.
// Grid: (B*H=64, T/64=16). Block: 128 threads = 4 waves; each wave owns 16
// query rows, streams 64-key tiles. S=Q*K^T (WMMA bf16), mask+scale, online
// softmax in f32 registers, P transposed through LDS into A-fragment layout,
// O += P*V (WMMA bf16 with V pre-transposed [B,H,HD,T]).
// ---------------------------------------------------------------------------
__global__ void __launch_bounds__(128)
attn_kernel(const __hip_bfloat16* __restrict__ Qb,
            const __hip_bfloat16* __restrict__ Kb,
            const __hip_bfloat16* __restrict__ Vtb,
            const float* __restrict__ adj,
            __hip_bfloat16* __restrict__ ctx) {
    constexpr int T = 1024, HD = 64, LDP = 72;   // LDP*2B = 144B, 16B-aligned
    __shared__ alignas(16) __hip_bfloat16 sP[4][16 * LDP];

    const int bh   = blockIdx.x;
    const int b    = bh >> 4;
    const int h    = bh & 15;
    const int tid  = threadIdx.x;
    const int lane = tid & 31;
    const int wave = tid >> 5;                   // 0..3
    const int lm   = lane & 15;
    const int hs   = lane >> 4;
    const int q0   = blockIdx.y * 64 + wave * 16;

    // Q fragments (A layout), resident for the whole key loop
    BFrag qf[2];
#pragma unroll
    for (int c = 0; c < 2; ++c) {
        const __hip_bfloat16* p =
            Qb + ((size_t)bh * T + q0 + lm) * HD + c * 32 + hs * 8;
        qf[c].u[0] = *(const v4u*)p;
        qf[c].u[1] = *(const v4u*)(p + 16);
    }

    const v8f vzero = {0.f, 0.f, 0.f, 0.f, 0.f, 0.f, 0.f, 0.f};
    v8f   o[4];
    float mrow[8], lrow[8];
#pragma unroll
    for (int dt = 0; dt < 4; ++dt) o[dt] = vzero;
#pragma unroll
    for (int r = 0; r < 8; ++r) { mrow[r] = -INFINITY; lrow[r] = 0.f; }

    for (int kt = 0; kt < T; kt += 64) {
        // S = Q * K^T  (contraction over HD=64 -> 2 chunks of 32)
        v8f s[4];
#pragma unroll
        for (int nt = 0; nt < 4; ++nt) s[nt] = vzero;
#pragma unroll
        for (int nt = 0; nt < 4; ++nt)
#pragma unroll
            for (int c = 0; c < 2; ++c) {
                const __hip_bfloat16* p =
                    Kb + ((size_t)bh * T + kt + nt * 16 + lm) * HD + c * 32 + hs * 8;
                BFrag kf;
                kf.u[0] = *(const v4u*)p;
                kf.u[1] = *(const v4u*)(p + 16);
                s[nt] = __builtin_amdgcn_wmma_f32_16x16x32_bf16(
                    false, qf[c].v, false, kf.v, (short)0, s[nt], false, false);
            }

        // scale by 1/sqrt(64) then adjacency mask (<=0 -> -1e30), as reference
#pragma unroll
        for (int nt = 0; nt < 4; ++nt)
#pragma unroll
            for (int r = 0; r < 8; ++r) {
                const int qg = q0 + r + 8 * hs;
                const int kg = kt + nt * 16 + lm;
                float v = s[nt][r] * 0.125f;
                if (adj[(size_t)qg * T + kg] <= 0.0f) v = -1.0e30f;
                s[nt][r] = v;
            }

        // online softmax per row r (row spans 16 lanes of one half-wave)
#pragma unroll
        for (int r = 0; r < 8; ++r) {
            float rm = fmaxf(fmaxf(s[0][r], s[1][r]), fmaxf(s[2][r], s[3][r]));
            rm = fmaxf(rm, __shfl_xor(rm, 1, 32));
            rm = fmaxf(rm, __shfl_xor(rm, 2, 32));
            rm = fmaxf(rm, __shfl_xor(rm, 4, 32));
            rm = fmaxf(rm, __shfl_xor(rm, 8, 32));
            const float mnew  = fmaxf(mrow[r], rm);
            const float alpha = __expf(mrow[r] - mnew);
            mrow[r] = mnew;
            float rs = 0.f;
#pragma unroll
            for (int nt = 0; nt < 4; ++nt) {
                const float pe = __expf(s[nt][r] - mnew);
                s[nt][r] = pe;
                rs += pe;
            }
            rs += __shfl_xor(rs, 1, 32);
            rs += __shfl_xor(rs, 2, 32);
            rs += __shfl_xor(rs, 4, 32);
            rs += __shfl_xor(rs, 8, 32);
            lrow[r] = lrow[r] * alpha + rs;
#pragma unroll
            for (int dt = 0; dt < 4; ++dt) o[dt][r] *= alpha;
        }

        // P (C layout) -> LDS bf16, then reload in A-fragment layout
#pragma unroll
        for (int nt = 0; nt < 4; ++nt)
#pragma unroll
            for (int r = 0; r < 8; ++r)
                sP[wave][(r + 8 * hs) * LDP + nt * 16 + lm] =
                    __float2bfloat16(s[nt][r]);
        __syncthreads();

        BFrag pf[2];
#pragma unroll
        for (int c = 0; c < 2; ++c) {
            const __hip_bfloat16* p = &sP[wave][lm * LDP + c * 32 + hs * 8];
            pf[c].u[0] = *(const v4u*)p;
            pf[c].u[1] = *(const v4u*)(p + 16);
        }

        // O += P * V  (contraction over 64 keys -> 2 chunks; V is [HD,T])
#pragma unroll
        for (int dt = 0; dt < 4; ++dt)
#pragma unroll
            for (int c = 0; c < 2; ++c) {
                const __hip_bfloat16* p =
                    Vtb + ((size_t)bh * HD + dt * 16 + lm) * T + kt + c * 32 + hs * 8;
                BFrag vf;
                vf.u[0] = *(const v4u*)p;
                vf.u[1] = *(const v4u*)(p + 16);
                o[dt] = __builtin_amdgcn_wmma_f32_16x16x32_bf16(
                    false, pf[c].v, false, vf.v, (short)0, o[dt], false, false);
            }
        __syncthreads();
    }

    // normalize and emit context [B*T, D] bf16 (input to out-proj GEMM)
#pragma unroll
    for (int dt = 0; dt < 4; ++dt)
#pragma unroll
        for (int r = 0; r < 8; ++r) {
            const int qg = q0 + r + 8 * hs;
            const int d  = dt * 16 + lm;
            const float v = o[dt][r] / lrow[r];
            ctx[((size_t)(b * 1024 + qg)) * 1024 + h * 64 + d] =
                __float2bfloat16(v);
        }
}

// ---------------------------------------------------------------------------
// Launch: convert -> QKV GEMM -> flash attention -> out-proj GEMM
// ---------------------------------------------------------------------------
extern "C" void kernel_launch(void* const* d_in, const int* in_sizes, int n_in,
                              void* d_out, int out_size, void* d_ws, size_t ws_size,
                              hipStream_t stream) {
    const float* x   = (const float*)d_in[0];   // [4,1024,1024]
    const float* adj = (const float*)d_in[1];   // [1024,1024]
    const float* w1  = (const float*)d_in[2];   // [3072,1024]
    const float* b1  = (const float*)d_in[3];   // [3072]
    const float* w2  = (const float*)d_in[4];   // [1024,1024]
    const float* b2  = (const float*)d_in[5];   // [1024]
    float* out = (float*)d_out;                 // [4,1024,1024]

    const size_t NX = (size_t)4096 * 1024;      // B*T*D elements
    __hip_bfloat16* xb  = (__hip_bfloat16*)d_ws;            // x in bf16
    __hip_bfloat16* w1b = xb  + NX;                         // 3072*1024
    __hip_bfloat16* w2b = w1b + (size_t)3072 * 1024;        // 1024*1024
    __hip_bfloat16* Qb  = w2b + (size_t)1024 * 1024;        // [B,H,T,HD]
    __hip_bfloat16* Kb  = Qb  + NX;                         // [B,H,T,HD]
    __hip_bfloat16* Vtb = Kb  + NX;                         // [B,H,HD,T]
    __hip_bfloat16* ctx = xb;   // reuse: x consumed by GEMM1 before attn writes

    f32_to_bf16_kernel<<<(int)(NX / 256), 256, 0, stream>>>(x, xb, (int)NX);
    f32_to_bf16_kernel<<<(3072 * 1024) / 256, 256, 0, stream>>>(w1, w1b, 3072 * 1024);
    f32_to_bf16_kernel<<<(1024 * 1024) / 256, 256, 0, stream>>>(w2, w2b, 1024 * 1024);

    gemm_bf16_kernel<0><<<dim3(32, 24), 256, 0, stream>>>(
        xb, w1b, b1, nullptr, Qb, Kb, Vtb);

    attn_kernel<<<dim3(64, 16), 128, 0, stream>>>(Qb, Kb, Vtb, adj, ctx);

    gemm_bf16_kernel<1><<<dim3(32, 8), 256, 0, stream>>>(
        ctx, w2b, b2, out, nullptr, nullptr, nullptr);
}
